// NonLocalAttention_80771154969262
// MI455X (gfx1250) — matrix operand order repaired and checked by reading, not compiled, not run on hardware
//
#include <hip/hip_runtime.h>
#include <hip/hip_bf16.h>
#include <math.h>

// ---------------------------------------------------------------------------
// Non-local attention for MI455X (gfx1250, wave32, WMMA).
// Fused flash-attention style: never materializes the 9216x9216 score matrix.
// Matrix cores: v_wmma_f32_16x16x32_f16 (K=32 == CR exactly).
//  - 64-key chunks: 4 score WMMAs + 2x5 PV WMMAs per iteration
//  - softmax row-sum folded into the PV GEMM via a "ones" channel tile
//  - row-max via packed-f16 butterfly (safe: any near-max m keeps exp exact)
//  - scores in log2 domain (Q pre-scaled by log2(e)) -> raw v_exp_f32
//  - K B-tiles register double-buffered; V A-tiles issued at iteration top
// ---------------------------------------------------------------------------

#define NB   2
#define CH   64
#define CRD  32
#define HWN  9216            // 96*96
#define QT   16              // queries per wave-tile
#define KC   64              // keys per inner chunk (two 32-key halves)

typedef __attribute__((ext_vector_type(16))) _Float16 v16h;
typedef __attribute__((ext_vector_type(8)))  _Float16 v8h;
typedef __attribute__((ext_vector_type(2)))  _Float16 v2h;
typedef __attribute__((ext_vector_type(8)))  float    v8f;
typedef __attribute__((ext_vector_type(4)))  float    v4f;

__device__ __forceinline__ v8f wmma16x16x32(v16h a, v16h b, v8f c) {
    // (neg_a, A, neg_b, B, c_mod, C, reuse_a, reuse_b)
    return __builtin_amdgcn_wmma_f32_16x16x32_f16(false, a, false, b,
                                                  (short)0, c, false, false);
}

__device__ __forceinline__ float fast_exp2(float x) {
#if __has_builtin(__builtin_amdgcn_exp2f)
    return __builtin_amdgcn_exp2f(x);     // v_exp_f32 (2^x is the native op)
#else
    return exp2f(x);
#endif
}

#define LOG2E 1.4426950408889634f

// ---------------------------------------------------------------------------
// Kernel 1: q/k/v 1x1-conv + ReLU projections, emitted as f16.
//   Qh,Kh : [N][HW][CR]  (pixel-major rows -> direct A/B tile loads)
//   Vh    : [N][C][HW]   (channel-major -> contiguous V^T tile rows)
// Q is additionally scaled by log2(e) so attention exps are pure exp2.
// ---------------------------------------------------------------------------
__global__ __launch_bounds__(256) void proj_kernel(
    const float* __restrict__ x,  const float* __restrict__ fm,
    const float* __restrict__ wq, const float* __restrict__ bq,
    const float* __restrict__ wk, const float* __restrict__ bk,
    const float* __restrict__ wa, const float* __restrict__ ba,
    _Float16* __restrict__ Qh, _Float16* __restrict__ Kh,
    _Float16* __restrict__ Vh)
{
    int p = blockIdx.x * 256 + threadIdx.x;
    if (p >= NB * HWN) return;
    int n = p / HWN, pix = p % HWN;
    const float* xn = x  + (size_t)n * CH * HWN + pix;
    const float* fn = fm + (size_t)n * CH * HWN + pix;

    float acc[CRD];

    // ---- q = relu(wq . x) * log2(e) ----
    #pragma unroll 4
    for (int o = 0; o < CRD; ++o) acc[o] = bq[o];
    for (int c = 0; c < CH; ++c) {
        float xv = xn[(size_t)c * HWN];
        #pragma unroll
        for (int o = 0; o < CRD; ++o) acc[o] += wq[o * CH + c] * xv;  // uniform -> s_load
    }
    {
        v16h q0, q1;
        #pragma unroll
        for (int o = 0; o < 16; ++o) q0[o] = (_Float16)(fmaxf(acc[o], 0.f) * LOG2E);
        #pragma unroll
        for (int o = 0; o < 16; ++o) q1[o] = (_Float16)(fmaxf(acc[16 + o], 0.f) * LOG2E);
        _Float16* qr = Qh + (size_t)(n * HWN + pix) * CRD;
        *(v16h*)qr = q0; *(v16h*)(qr + 16) = q1;
    }

    // ---- k = relu(wk . fusionmap) ----
    #pragma unroll 4
    for (int o = 0; o < CRD; ++o) acc[o] = bk[o];
    for (int c = 0; c < CH; ++c) {
        float fv = fn[(size_t)c * HWN];
        #pragma unroll
        for (int o = 0; o < CRD; ++o) acc[o] += wk[o * CH + c] * fv;
    }
    {
        v16h k0, k1;
        #pragma unroll
        for (int o = 0; o < 16; ++o) k0[o] = (_Float16)fmaxf(acc[o], 0.f);
        #pragma unroll
        for (int o = 0; o < 16; ++o) k1[o] = (_Float16)fmaxf(acc[16 + o], 0.f);
        _Float16* kr = Kh + (size_t)(n * HWN + pix) * CRD;
        *(v16h*)kr = k0; *(v16h*)(kr + 16) = k1;
    }

    // ---- v = relu(wa . fusionmap), channel-major, two halves of 32 ----
    for (int half = 0; half < 2; ++half) {
        #pragma unroll 4
        for (int o = 0; o < 32; ++o) acc[o] = ba[half * 32 + o];
        for (int c = 0; c < CH; ++c) {
            float fv = fn[(size_t)c * HWN];
            #pragma unroll
            for (int o = 0; o < 32; ++o) acc[o] += wa[(half * 32 + o) * CH + c] * fv;
        }
        #pragma unroll 4
        for (int o = 0; o < 32; ++o)
            Vh[(size_t)(n * CH + half * 32 + o) * HWN + pix] =
                (_Float16)fmaxf(acc[o], 0.f);
    }
}

// ---------------------------------------------------------------------------
// Kernel 2: fused attention (out^T formulation), 1 wave per 16-query tile.
//   out^T[c][q] = sum_k V^T[c][k] * softmax_row(q)[k]
//
// Layout (ISA 7.12.2, wave32):
//   f16 A 16x32 : lane L row M=L&15; elem j<8 -> K=hi*8+j, j>=8 -> 16+hi*8+(j-8)
//   f16 B 32x16 : lane L col N=L&15; code uses K=16*hi+j (contiguous) for both A and B
//   f32 C/D     : lane L col N=L&15; VGPR r -> row M = r + 8*(L>>4)
//
// 64-key chunk: score tiles 0/1 cover even/odd keys of [kb,kb+32), tiles 2/3
// of [kb+32,kb+64). A lane's tile-pair probabilities land at adjacent B
// positions (2c,2c+1) -> packed b32 LDS stores, while B(elem j)->key kb+j
// stays the identity so V A-tile loads are contiguous b128.
// Denominator = extra PV WMMA with a ones-channel A tile (row 0 = 1).
// ---------------------------------------------------------------------------
__global__ __launch_bounds__(32) void attn_kernel(
    const _Float16* __restrict__ Qh, const _Float16* __restrict__ Kh,
    const _Float16* __restrict__ Vh, float* __restrict__ Out)
{
    __shared__ __align__(64) _Float16 Pld[QT * KC];   // P in [q][k], rows 128B
    __shared__ __align__(64) float corrLd[QT];

    const int lane = threadIdx.x;
    const int col  = lane & 15;
    const int hi   = lane >> 4;
    const int tile = blockIdx.x;               // 0 .. NB*(HW/16)-1
    const int n    = tile / (HWN / QT);
    const int q0   = (tile % (HWN / QT)) * QT;

    const _Float16* Qn = Qh + (size_t)n * HWN * CRD;
    const _Float16* Kn = Kh + (size_t)n * HWN * CRD;
    const _Float16* Vn = Vh + (size_t)n * CH * HWN;

    // Q A-tile: invariant over the whole key loop.
    v16h aq;
    {
        const _Float16* qrow = Qn + (size_t)(q0 + col) * CRD;
        v8h lo = *(const v8h*)(qrow + hi * 16);
        v8h h8 = *(const v8h*)(qrow + hi * 16 + 8);
        #pragma unroll
        for (int j = 0; j < 8; ++j) { aq[j] = lo[j]; aq[8 + j] = h8[j]; }
    }
    // "ones channel" A-tile for the denominator GEMM: row 0 = 1, rest = 0.
    v16h aone;
    {
        _Float16 one = (_Float16)((col == 0) ? 1.0f : 0.0f);
        #pragma unroll
        for (int j = 0; j < 16; ++j) aone[j] = one;
    }

    v8f acc[4], accL;
    #pragma unroll
    for (int t = 0; t < 4; ++t) acc[t] = (v8f){};
    accL = (v8f){};
    float m_prev[8];
    #pragma unroll
    for (int r = 0; r < 8; ++r) m_prev[r] = -1e30f;

    // K B-tile rows for a chunk base: t0 even / t1 odd of first 32 keys,
    // t2 even / t3 odd of second 32 keys.
    v16h bk[4];
    #pragma unroll
    for (int t = 0; t < 4; ++t) {
        int row = (t >> 1) * 32 + 2 * col + (t & 1);
        const _Float16* kr = Kn + (size_t)row * CRD + hi * 16;
        v8h a = *(const v8h*)kr, b = *(const v8h*)(kr + 8);
        #pragma unroll
        for (int j = 0; j < 8; ++j) { bk[t][j] = a[j]; bk[t][8 + j] = b[j]; }
    }

    for (int kb = 0; kb < HWN; kb += KC) {
        const int kbn = (kb + KC == HWN) ? 0 : kb + KC;   // wrapped: always valid

        // V A-tiles for THIS chunk: issue all loads now, consume after softmax
        v16h av[4][2];
        #pragma unroll
        for (int t = 0; t < 4; ++t)
            #pragma unroll
            for (int h = 0; h < 2; ++h) {
                const _Float16* vrow =
                    Vn + (size_t)(t * 16 + col) * HWN + kb + h * 32;
                v8h lo = *(const v8h*)(vrow + hi * 16);
                v8h h8 = *(const v8h*)(vrow + hi * 16 + 8);
                #pragma unroll
                for (int j = 0; j < 8; ++j) { av[t][h][j] = lo[j]; av[t][h][8 + j] = h8[j]; }
            }
        // prefetch next chunk's V rows (2 rows/lane, 128B spans the 64 keys)
        __builtin_prefetch(Vn + (size_t)(lane * 2) * HWN + kbn, 0, 3);
        __builtin_prefetch(Vn + (size_t)(lane * 2 + 1) * HWN + kbn, 0, 3);

        // scores for this chunk from double-buffered K tiles (log2 units)
        v8f z = (v8f){};
        v8f s0 = wmma16x16x32(aq, bk[0], z);
        v8f s1 = wmma16x16x32(aq, bk[1], z);
        v8f s2 = wmma16x16x32(aq, bk[2], z);
        v8f s3 = wmma16x16x32(aq, bk[3], z);

        // issue NEXT chunk's K loads (overlap with softmax + PV)
        v16h bkn[4];
        #pragma unroll
        for (int t = 0; t < 4; ++t) {
            int row = kbn + (t >> 1) * 32 + 2 * col + (t & 1);
            const _Float16* kr = Kn + (size_t)row * CRD + hi * 16;
            v8h a = *(const v8h*)kr, b = *(const v8h*)(kr + 8);
            #pragma unroll
            for (int j = 0; j < 8; ++j) { bkn[t][j] = a[j]; bkn[t][8 + j] = b[j]; }
        }

        // ---- packed-f16 row-max butterfly over all 64 chunk keys ----
        float mn[8], corr[8];
        #pragma unroll
        for (int i = 0; i < 4; ++i) {
            v2h pk;
            pk[0] = (_Float16)fmaxf(fmaxf(s0[2 * i],     s1[2 * i]),
                                    fmaxf(s2[2 * i],     s3[2 * i]));
            pk[1] = (_Float16)fmaxf(fmaxf(s0[2 * i + 1], s1[2 * i + 1]),
                                    fmaxf(s2[2 * i + 1], s3[2 * i + 1]));
            #pragma unroll
            for (int msk = 1; msk <= 8; msk <<= 1) {
                int w = __shfl_xor(__builtin_bit_cast(int, pk), msk, 32);
                v2h o = __builtin_bit_cast(v2h, w);
                pk[0] = (pk[0] > o[0]) ? pk[0] : o[0];
                pk[1] = (pk[1] > o[1]) ? pk[1] : o[1];
            }
            mn[2 * i]     = fmaxf(m_prev[2 * i],     (float)pk[0]);
            mn[2 * i + 1] = fmaxf(m_prev[2 * i + 1], (float)pk[1]);
        }
        // ---- exp2 + packed P scatter (two b32 stores per row) ----
        #pragma unroll
        for (int r = 0; r < 8; ++r) {
            corr[r] = fast_exp2(m_prev[r] - mn[r]);
            m_prev[r] = mn[r];
            float p0 = fast_exp2(s0[r] - mn[r]);
            float p1 = fast_exp2(s1[r] - mn[r]);
            float p2 = fast_exp2(s2[r] - mn[r]);
            float p3 = fast_exp2(s3[r] - mn[r]);
            v2h pa; pa[0] = (_Float16)p0; pa[1] = (_Float16)p1;
            v2h pb; pb[0] = (_Float16)p2; pb[1] = (_Float16)p3;
            _Float16* pr = Pld + (r + 8 * hi) * KC + 2 * col;
            *(v2h*)pr        = pa;
            *(v2h*)(pr + 32) = pb;
        }
        // broadcast per-query correction to accumulator columns (2 x b128)
        if (col == 0) {
            v4f c0, c1;
            #pragma unroll
            for (int i = 0; i < 4; ++i) { c0[i] = corr[i]; c1[i] = corr[4 + i]; }
            *(v4f*)(corrLd + 8 * hi)     = c0;
            *(v4f*)(corrLd + 8 * hi + 4) = c1;
        }
        __syncthreads();

        float cfac = corrLd[col];
        #pragma unroll
        for (int t = 0; t < 4; ++t)
            #pragma unroll
            for (int r = 0; r < 8; ++r) acc[t][r] *= cfac;
        #pragma unroll
        for (int r = 0; r < 8; ++r) accL[r] *= cfac;

        // P^T B-tiles from LDS (lane col = query, lane-half splits keys)
        v16h bp0, bp1;
        {
            const _Float16* pr = Pld + col * KC + hi * 16;
            v8h a = *(const v8h*)pr,        b = *(const v8h*)(pr + 8);
            v8h c = *(const v8h*)(pr + 32), d = *(const v8h*)(pr + 40);
            #pragma unroll
            for (int j = 0; j < 8; ++j) {
                bp0[j] = a[j]; bp0[8 + j] = b[j];
                bp1[j] = c[j]; bp1[8 + j] = d[j];
            }
        }

        // PV: 4 channel-tiles + ones-row denominator, both 32-key halves
        #pragma unroll
        for (int t = 0; t < 4; ++t) acc[t] = wmma16x16x32(av[t][0], bp0, acc[t]);
        accL = wmma16x16x32(aone, bp0, accL);
        #pragma unroll
        for (int t = 0; t < 4; ++t) acc[t] = wmma16x16x32(av[t][1], bp1, acc[t]);
        accL = wmma16x16x32(aone, bp1, accL);

        // rotate K double-buffer
        #pragma unroll
        for (int t = 0; t < 4; ++t) bk[t] = bkn[t];

        __syncthreads();   // protect Pld against next chunk's writes
    }

    // finalize: l for query q sits in accL row 0 (lane q, reg 0)
    float l = __shfl(accL[0], col, 32);
    float linv = 1.0f / l;
    float* On = Out + (size_t)n * CH * HWN;
    #pragma unroll
    for (int t = 0; t < 4; ++t)
        #pragma unroll
        for (int r = 0; r < 8; ++r)
            On[(size_t)(t * 16 + r + 8 * hi) * HWN + q0 + col] = acc[t][r] * linv;
}

// ---------------------------------------------------------------------------
// Kernel 3: out = relu(wo . concat(x, attn) + bo)
// ---------------------------------------------------------------------------
__global__ __launch_bounds__(256) void out_conv_kernel(
    const float* __restrict__ x, const float* __restrict__ attn,
    const float* __restrict__ wo, const float* __restrict__ bo,
    float* __restrict__ out)
{
    int p = blockIdx.x * 256 + threadIdx.x;
    if (p >= NB * HWN) return;
    int n = p / HWN, pix = p % HWN;
    const float* xn = x    + (size_t)n * CH * HWN + pix;
    const float* an = attn + (size_t)n * CH * HWN + pix;
    float* on       = out  + (size_t)n * CH * HWN + pix;

    for (int og = 0; og < 4; ++og) {
        float acc[16];
        #pragma unroll
        for (int o = 0; o < 16; ++o) acc[o] = bo[og * 16 + o];
        for (int c = 0; c < CH; ++c) {
            float xv = xn[(size_t)c * HWN];
            #pragma unroll
            for (int o = 0; o < 16; ++o)
                acc[o] += wo[(og * 16 + o) * (2 * CH) + c] * xv;
        }
        for (int c = 0; c < CH; ++c) {
            float av = an[(size_t)c * HWN];
            #pragma unroll
            for (int o = 0; o < 16; ++o)
                acc[o] += wo[(og * 16 + o) * (2 * CH) + CH + c] * av;
        }
        #pragma unroll
        for (int o = 0; o < 16; ++o)
            on[(size_t)(og * 16 + o) * HWN] = fmaxf(acc[o], 0.f);
    }
}

// ---------------------------------------------------------------------------
extern "C" void kernel_launch(void* const* d_in, const int* in_sizes, int n_in,
                              void* d_out, int out_size, void* d_ws, size_t ws_size,
                              hipStream_t stream) {
    const float* x  = (const float*)d_in[0];
    const float* fm = (const float*)d_in[1];
    const float* wq = (const float*)d_in[2];
    const float* bq = (const float*)d_in[3];
    const float* wk = (const float*)d_in[4];
    const float* bk = (const float*)d_in[5];
    const float* wa = (const float*)d_in[6];
    const float* ba = (const float*)d_in[7];
    const float* wo = (const float*)d_in[8];
    const float* bo = (const float*)d_in[9];
    float* out = (float*)d_out;

    // workspace: Qh(1.18MB) Kh(1.18MB) Vh(2.36MB) Attn f32(4.72MB) ~ 9.4MB
    _Float16* Qh = (_Float16*)d_ws;
    _Float16* Kh = Qh + (size_t)NB * HWN * CRD;
    _Float16* Vh = Kh + (size_t)NB * HWN * CRD;
    float*  Attn = (float*)(Vh + (size_t)NB * CH * HWN);

    int npix = NB * HWN;
    proj_kernel<<<(npix + 255) / 256, 256, 0, stream>>>(
        x, fm, wq, bq, wk, bk, wa, ba, Qh, Kh, Vh);

    attn_kernel<<<NB * (HWN / QT), 32, 0, stream>>>(Qh, Kh, Vh, Attn);

    out_conv_kernel<<<(npix + 255) / 256, 256, 0, stream>>>(x, Attn, wo, bo, out);
}
